// NonLocalBlock2D_with_mask_Res_16338055594200
// MI455X (gfx1250) — matrix operand order repaired and checked by reading, not compiled, not run on hardware
//
#include <hip/hip_runtime.h>
#include <hip/hip_bf16.h>
#include <stdint.h>

typedef __bf16 bf16;
typedef __attribute__((ext_vector_type(16))) __bf16 v16bf;
typedef __attribute__((ext_vector_type(2)))  __bf16 v2bf;
typedef __attribute__((ext_vector_type(8)))  float  v8f;

#define B_ 4
#define C_ 256
#define N_ 4096

// ---------------- bf16 helpers ----------------
#if __has_builtin(__builtin_amdgcn_cvt_pk_bf16_f32)
__device__ inline uint32_t pk2(float lo, float hi) {
  union { v2bf v; uint32_t u; } p;
  p.v = __builtin_amdgcn_cvt_pk_bf16_f32(lo, hi);   // v_cvt_pk_bf16_f32
  return p.u;
}
__device__ inline bf16 f2bf(float f) {
  union { v2bf v; uint32_t u; } p;
  p.v = __builtin_amdgcn_cvt_pk_bf16_f32(f, 0.f);
  return p.v[0];
}
#else
__device__ inline bf16 f2bf(float f) {
  union { float f; uint32_t u; } x; x.f = f;
  uint32_t r = x.u + 0x7FFFu + ((x.u >> 16) & 1u);   // round-to-nearest-even
  union { uint16_t s; bf16 b; } o; o.s = (uint16_t)(r >> 16);
  return o.b;
}
__device__ inline uint32_t pk2(float lo, float hi) {
  union { float f; uint32_t u; } a, b; a.f = lo; b.f = hi;
  uint32_t ra = a.u + 0x7FFFu + ((a.u >> 16) & 1u);
  uint32_t rb = b.u + 0x7FFFu + ((b.u >> 16) & 1u);
  return (ra >> 16) | (rb & 0xFFFF0000u);
}
#endif

union FragU { v16bf v; uint32_t u[8]; };

// A fragment (16x32 bf16): lane<16 -> row M=lane, K=0..15; lane>=16 -> K=16..31
__device__ inline v16bf frag_A(const bf16* base, int ld, int lane) {
  int row = lane & 15, half = lane >> 4;
  FragU f;
#pragma unroll
  for (int j = 0; j < 8; ++j) {
    int k = 2 * j + 16 * half;
    f.u[j] = *(const uint32_t*)(base + row * ld + k);
  }
  return f.v;
}

// B fragment from K-contiguous (transposed BT[col][k]) storage: lane<16 K=0..15, lane>=16 K=16..31
__device__ inline v16bf frag_BT(const bf16* base, int ld, int lane) {
  int col = lane & 15, half = lane >> 4;
  FragU f;
#pragma unroll
  for (int j = 0; j < 8; ++j) {
    int k = 2 * j + 16 * half;
    f.u[j] = *(const uint32_t*)(base + col * ld + k);
  }
  return f.v;
}

// B fragment from natural Bn[k][col] storage with the P-packing key permutation:
// P column k holds key (k>>1) + (k&1)*16, so gather G rows in that order.
__device__ inline v16bf frag_B_perm(const bf16* base, int ld, int lane) {
  int col = lane & 15, half = lane >> 4;
  v16bf f;
#pragma unroll
  for (int j = 0; j < 8; ++j) {
    int k = 2 * j + 16 * half;          // even
    int k0 = (k >> 1);                  // perm(k)
    int k1 = (k >> 1) + 16;             // perm(k+1)
    f[2 * j]     = base[k0 * ld + col];
    f[2 * j + 1] = base[k1 * ld + col];
  }
  return f;
}

// ---------------- mask pipeline ----------------
__global__ void mask_k(const float* __restrict__ mask, float* __restrict__ Mf) {
  int idx = blockIdx.x * blockDim.x + threadIdx.x;
  if (idx >= B_ * N_) return;
  int b = idx / N_, p = idx % N_;
  int hh = p / 64, ww = p % 64;
  const float* mb = mask + b * 32 * 32;
  float sh = (hh + 0.5f) * 0.5f - 0.5f;
  float sw = (ww + 0.5f) * 0.5f - 0.5f;
  int h0f = (int)floorf(sh), w0f = (int)floorf(sw);
  float fh = sh - (float)h0f, fw = sw - (float)w0f;
  int h0 = min(max(h0f, 0), 31), h1 = min(max(h0f + 1, 0), 31);
  int w0 = min(max(w0f, 0), 31), w1 = min(max(w0f + 1, 0), 31);
  float bil = mb[h0 * 32 + w0] * (1.f - fh) * (1.f - fw)
            + mb[h0 * 32 + w1] * (1.f - fh) * fw
            + mb[h1 * 32 + w0] * fh * (1.f - fw)
            + mb[h1 * 32 + w1] * fh * fw;
  float m = bil > 0.f ? 1.f : bil;
  m = 1.f - m;
  float tmp = 1.f - mb[(hh >> 1) * 32 + (ww >> 1)];   // nearest upsample of (1-mask)
  Mf[idx] = m * tmp;
}

// ---------------- 1x1 conv (theta/phi/g) as WMMA GEMM, bf16 out [B,N,C] ----------------
__global__ void __launch_bounds__(128)
gemm_qkv(const float* __restrict__ x,
         const float* __restrict__ wq, const float* __restrict__ bq,
         const float* __restrict__ wk, const float* __restrict__ bk,
         const float* __restrict__ wv, const float* __restrict__ bv,
         bf16* __restrict__ oq, bf16* __restrict__ ok, bf16* __restrict__ ov) {
  __shared__ bf16 aT[64][40];   // [n][c]
  __shared__ bf16 bT[32][32];   // [o][c]
  int tid = threadIdx.x, lane = tid & 31, w = tid >> 5;
  int n0 = blockIdx.x * 64, o0 = blockIdx.y * 32;
  int z = blockIdx.z; int b = z / 3, which = z % 3;
  const float* wm = which == 0 ? wq : (which == 1 ? wk : wv);
  const float* bm = which == 0 ? bq : (which == 1 ? bk : bv);
  bf16* om = which == 0 ? oq : (which == 1 ? ok : ov);

  v8f acc0 = {}, acc1 = {};
  const float* xb = x + (size_t)b * C_ * N_;

  for (int kc = 0; kc < C_; kc += 32) {
    { // A: aT[n][cp..cp+1] = x[b][kc+cp..][n0+n]; each thread owns a channel pair -> b32 stores
      int cp = (tid & 15) * 2, nb = (tid >> 4) * 8;
      const float* s0 = xb + (size_t)(kc + cp) * N_ + n0 + nb;
      const float* s1 = s0 + N_;
#pragma unroll
      for (int i = 0; i < 8; i += 4) {
        float4 a0 = *(const float4*)(s0 + i);
        float4 a1 = *(const float4*)(s1 + i);
        *(uint32_t*)&aT[nb + i + 0][cp] = pk2(a0.x, a1.x);
        *(uint32_t*)&aT[nb + i + 1][cp] = pk2(a0.y, a1.y);
        *(uint32_t*)&aT[nb + i + 2][cp] = pk2(a0.z, a1.z);
        *(uint32_t*)&aT[nb + i + 3][cp] = pk2(a0.w, a1.w);
      }
    }
    { // B: bT[o][c] = w[o0+o][kc+c] (packed pair stores)
      int o = tid >> 2, cg = (tid & 3) * 8;
      const float* src = wm + (size_t)(o0 + o) * C_ + kc + cg;
      float4 v0 = *(const float4*)(src);
      float4 v1 = *(const float4*)(src + 4);
      *(uint32_t*)&bT[o][cg + 0] = pk2(v0.x, v0.y);
      *(uint32_t*)&bT[o][cg + 2] = pk2(v0.z, v0.w);
      *(uint32_t*)&bT[o][cg + 4] = pk2(v1.x, v1.y);
      *(uint32_t*)&bT[o][cg + 6] = pk2(v1.z, v1.w);
    }
    __syncthreads();
    v16bf a  = frag_A(&aT[w * 16][0], 40, lane);
    v16bf b0 = frag_BT(&bT[0][0],  32, lane);
    v16bf b1 = frag_BT(&bT[16][0], 32, lane);
    acc0 = __builtin_amdgcn_wmma_f32_16x16x32_bf16(false, a, false, b0, (short)0, acc0, false, false);
    acc1 = __builtin_amdgcn_wmma_f32_16x16x32_bf16(false, a, false, b1, (short)0, acc1, false, false);
    __syncthreads();
  }
  int col = lane & 15, half = lane >> 4;
  float bias0 = bm[o0 + col], bias1 = bm[o0 + 16 + col];
#pragma unroll
  for (int r = 0; r < 8; ++r) {
    int M = r + 8 * half;
    size_t rowoff = ((size_t)b * N_ + n0 + w * 16 + M) * C_;
    om[rowoff + o0 + col]      = f2bf(acc0[r] + bias0);
    om[rowoff + o0 + 16 + col] = f2bf(acc1[r] + bias1);
  }
}

// ---------------- fused masked flash attention ----------------
__global__ void __launch_bounds__(128)
attn(const bf16* __restrict__ th, const bf16* __restrict__ ph, const bf16* __restrict__ gm,
     const float* __restrict__ maskv, bf16* __restrict__ yo) {
  __shared__ bf16 tl[64 * 264];      // theta [q][c]
  __shared__ bf16 pl[32 * 264];      // phi   [key][c]
  __shared__ bf16 gl[32 * 264];      // g     [key][c]
  __shared__ bf16 ps[4][16][36];     // P per wave, A layout (keys permuted: col 2n<-key n, 2n+1<-key n+16)
  __shared__ float ml[32];
  int tid = threadIdx.x, lane = tid & 31, w = tid >> 5;
  int q0 = blockIdx.x * 64, b = blockIdx.y;

  { // resident theta tile (64 x 256)
    int q = tid >> 1, cg = (tid & 1) * 128;
    const bf16* src = th + ((size_t)b * N_ + q0 + q) * C_ + cg;
#pragma unroll
    for (int i = 0; i < 128; i += 8)
      *(uint4*)&tl[q * 264 + cg + i] = *(const uint4*)(src + i);
  }
  float mx[8], ls[8];
  v8f Oa[16] = {};
#pragma unroll
  for (int r = 0; r < 8; ++r) { mx[r] = -1e30f; ls[r] = 0.f; }

  int n = lane & 15, half = lane >> 4;

  for (int kb = 0; kb < N_; kb += 32) {
    __syncthreads();
    { // stage key chunk (phi + g + mask)
      int key = tid >> 2, cg = (tid & 3) * 64;
      const bf16* sp = ph + ((size_t)b * N_ + kb + key) * C_ + cg;
      const bf16* sg = gm + ((size_t)b * N_ + kb + key) * C_ + cg;
#pragma unroll
      for (int i = 0; i < 64; i += 8) {
        *(uint4*)&pl[key * 264 + cg + i] = *(const uint4*)(sp + i);
        *(uint4*)&gl[key * 264 + cg + i] = *(const uint4*)(sg + i);
      }
      if (tid < 32) ml[tid] = maskv[(size_t)b * N_ + kb + tid];
      if (kb + 32 < N_)   // global_prefetch_b8 for next chunk
        __builtin_prefetch(ph + ((size_t)b * N_ + kb + 32 + key) * C_ + cg, 0, 0);
    }
    __syncthreads();

    // S = theta . phi^T  (two 16x16 D tiles per wave, K=256)
    v8f s0 = {}, s1 = {};
#pragma unroll
    for (int kc = 0; kc < 256; kc += 32) {
      v16bf a  = frag_A(&tl[(w * 16) * 264 + kc], 264, lane);
      v16bf b0 = frag_BT(&pl[kc],            264, lane);
      v16bf b1 = frag_BT(&pl[16 * 264 + kc], 264, lane);
      s0 = __builtin_amdgcn_wmma_f32_16x16x32_bf16(false, a, false, b0, (short)0, s0, false, false);
      s1 = __builtin_amdgcn_wmma_f32_16x16x32_bf16(false, a, false, b1, (short)0, s1, false, false);
    }

    // online masked softmax (row reductions across the 16-lane half-group owning each row)
    float mv0 = ml[n], mv1 = ml[n + 16];
#pragma unroll
    for (int r = 0; r < 8; ++r) {
      float rm = fmaxf(s0[r], s1[r]);
#pragma unroll
      for (int m2 = 1; m2 < 16; m2 <<= 1) rm = fmaxf(rm, __shfl_xor(rm, m2, 32));
      float nm = fmaxf(mx[r], rm);
      float sc = __expf(mx[r] - nm);
      float p0 = __expf(s0[r] - nm) * mv0;
      float p1 = __expf(s1[r] - nm) * mv1;
      float rs = p0 + p1;
#pragma unroll
      for (int m2 = 1; m2 < 16; m2 <<= 1) rs += __shfl_xor(rs, m2, 32);
      ls[r] = ls[r] * sc + rs;
      mx[r] = nm;
#pragma unroll
      for (int t = 0; t < 16; ++t) Oa[t][r] *= sc;
      // packed store: columns (2n, 2n+1) hold keys (n, n+16)
      *(uint32_t*)&ps[w][r + 8 * half][2 * n] = pk2(p0, p1);
    }

    // O += P . G  (16 C-subtiles); G rows gathered with matching key permutation
    v16bf ap = frag_A(&ps[w][0][0], 36, lane);
#pragma unroll
    for (int t = 0; t < 16; ++t) {
      v16bf bg = frag_B_perm(gl + t * 16, 264, lane);
      Oa[t] = __builtin_amdgcn_wmma_f32_16x16x32_bf16(false, ap, false, bg, (short)0, Oa[t], false, false);
    }
  }

  // y = O / l
#pragma unroll
  for (int r = 0; r < 8; ++r) {
    float inv = 1.f / fmaxf(ls[r], 1e-30f);
    size_t row = ((size_t)b * N_ + q0 + w * 16 + r + 8 * half) * C_;
#pragma unroll
    for (int t = 0; t < 16; ++t)
      yo[row + t * 16 + n] = f2bf(Oa[t][r] * inv);
  }
}

// ---------------- W 1x1 conv: y(bf16 [B,N,C]) x W -> fp32 [B,C,N] ----------------
__global__ void __launch_bounds__(128)
gemm_w(const bf16* __restrict__ y, const float* __restrict__ wm, const float* __restrict__ bm,
       float* __restrict__ out) {
  __shared__ bf16 aT[64][40];
  __shared__ bf16 bT[32][32];
  int tid = threadIdx.x, lane = tid & 31, w = tid >> 5;
  int n0 = blockIdx.x * 64, o0 = blockIdx.y * 32, b = blockIdx.z;
  v8f acc0 = {}, acc1 = {};
  for (int kc = 0; kc < C_; kc += 32) {
    { // A contiguous in c
      int n = tid >> 1, cg = (tid & 1) * 16;
      const bf16* src = y + ((size_t)b * N_ + n0 + n) * C_ + kc + cg;
      *(uint4*)&aT[n][cg]     = *(const uint4*)(src);
      *(uint4*)&aT[n][cg + 8] = *(const uint4*)(src + 8);
    }
    { // weights (packed pair stores)
      int o = tid >> 2, cg = (tid & 3) * 8;
      const float* src = wm + (size_t)(o0 + o) * C_ + kc + cg;
      float4 v0 = *(const float4*)(src);
      float4 v1 = *(const float4*)(src + 4);
      *(uint32_t*)&bT[o][cg + 0] = pk2(v0.x, v0.y);
      *(uint32_t*)&bT[o][cg + 2] = pk2(v0.z, v0.w);
      *(uint32_t*)&bT[o][cg + 4] = pk2(v1.x, v1.y);
      *(uint32_t*)&bT[o][cg + 6] = pk2(v1.z, v1.w);
    }
    __syncthreads();
    v16bf a  = frag_A(&aT[w * 16][0], 40, lane);
    v16bf b0 = frag_BT(&bT[0][0],  32, lane);
    v16bf b1 = frag_BT(&bT[16][0], 32, lane);
    acc0 = __builtin_amdgcn_wmma_f32_16x16x32_bf16(false, a, false, b0, (short)0, acc0, false, false);
    acc1 = __builtin_amdgcn_wmma_f32_16x16x32_bf16(false, a, false, b1, (short)0, acc1, false, false);
    __syncthreads();
  }
  int col = lane & 15, half = lane >> 4;
  float bias0 = bm[o0 + col], bias1 = bm[o0 + 16 + col];
#pragma unroll
  for (int r = 0; r < 8; ++r) {
    int M = r + 8 * half;
    out[((size_t)b * C_ + o0 + col)      * N_ + n0 + w * 16 + M] = acc0[r] + bias0;
    out[((size_t)b * C_ + o0 + 16 + col) * N_ + n0 + w * 16 + M] = acc1[r] + bias1;
  }
}

// ---------------- reflect-pad 3x3 conv as implicit-im2col WMMA GEMM ----------------
__global__ void __launch_bounds__(128)
conv3(const float* __restrict__ in, const float* __restrict__ prm, int donorm,
      const float* __restrict__ wt, const float* __restrict__ bias,
      float* __restrict__ out, float* __restrict__ stats) {
  __shared__ bf16 inT[3][36][32];   // [ky][pixel][c]
  __shared__ bf16 wl[9][32][32];    // [tap][o][c]
  int tid = threadIdx.x, lane = tid & 31, w = tid >> 5;
  int b = blockIdx.z, o0 = blockIdx.y * 32;
  int prow = blockIdx.x >> 1, pcol0 = (blockIdx.x & 1) * 32;
  int o_off = (w & 1) * 16, p_off = (w >> 1) * 16;
  v8f acc = {};
  const float* inb = in + (size_t)b * C_ * N_;

  for (int cc = 0; cc < C_; cc += 32) {
    { // input tile with fused IN/ReLU; thread owns a channel pair -> b32 stores
      int cp = (tid & 15) * 2, g = tid >> 4;   // 8 pixel-groups
      float mu0 = 0.f, rs0 = 1.f, mu1 = 0.f, rs1 = 1.f;
      if (donorm) {
        mu0 = prm[b * C_ + cc + cp];     rs0 = prm[B_ * C_ + b * C_ + cc + cp];
        mu1 = prm[b * C_ + cc + cp + 1]; rs1 = prm[B_ * C_ + b * C_ + cc + cp + 1];
      }
      const float* s0 = inb + (size_t)(cc + cp) * N_;
      const float* s1 = s0 + N_;
      for (int lc = g; lc < 34; lc += 8) {
        int scol = pcol0 - 1 + lc;
        scol = scol < 0 ? -scol : (scol > 63 ? 126 - scol : scol);
#pragma unroll
        for (int rr = 0; rr < 3; ++rr) {
          int srow = prow + rr - 1;
          srow = srow < 0 ? -srow : (srow > 63 ? 126 - srow : srow);
          float va = s0[srow * 64 + scol];
          float vb = s1[srow * 64 + scol];
          if (donorm) {
            va = fmaxf((va - mu0) * rs0, 0.f);
            vb = fmaxf((vb - mu1) * rs1, 0.f);
          }
          *(uint32_t*)&inT[rr][lc][cp] = pk2(va, vb);
        }
      }
    }
    { // weights wl[tap][o][c]; two adjacent channels are 18 consecutive floats
      int o = tid & 31, tg = tid >> 5;
      for (int cp = tg; cp < 16; cp += 4) {
        const float* src = wt + ((size_t)(o0 + o) * C_ + cc + cp * 2) * 9;
#pragma unroll
        for (int t9 = 0; t9 < 9; ++t9)
          *(uint32_t*)&wl[t9][o][cp * 2] = pk2(src[t9], src[9 + t9]);
      }
    }
    __syncthreads();
#pragma unroll
    for (int t9 = 0; t9 < 9; ++t9) {
      int ky = t9 / 3, kx = t9 % 3;
      v16bf a  = frag_A(&wl[t9][o_off][0], 32, lane);              // rows = o, K = c
      v16bf bi = frag_BT(&inT[ky][p_off + kx][0], 32, lane);       // cols = pixel, K = c
      acc = __builtin_amdgcn_wmma_f32_16x16x32_bf16(false, a, false, bi, (short)0, acc, false, false);
    }
    __syncthreads();
  }
  int nol = lane & 15, half = lane >> 4;
#pragma unroll
  for (int r = 0; r < 8; ++r) {
    int o = o0 + o_off + r + 8 * half;
    float v = acc[r] + bias[o];
    int pix = prow * 64 + pcol0 + p_off + nol;
    out[((size_t)b * C_ + o) * N_ + pix] = v;
    float s = v, s2 = v * v;
#pragma unroll
    for (int m2 = 1; m2 < 16; m2 <<= 1) { s += __shfl_xor(s, m2, 32); s2 += __shfl_xor(s2, m2, 32); }
    if (nol == 0) {
      atomicAdd(&stats[(b * C_ + o) * 2 + 0], s);
      atomicAdd(&stats[(b * C_ + o) * 2 + 1], s2);
    }
  }
}

// ---------------- small elementwise kernels ----------------
__global__ void zero_k(float* __restrict__ p, int nelem) {
  int i = blockIdx.x * blockDim.x + threadIdx.x;
  if (i < nelem) p[i] = 0.f;
}
__global__ void finalize_k(const float* __restrict__ stats, float* __restrict__ prm) {
  int i = blockIdx.x * blockDim.x + threadIdx.x;
  if (i >= B_ * C_) return;
  float s = stats[2 * i], s2 = stats[2 * i + 1];
  float mu = s / (float)N_;
  float var = s2 / (float)N_ - mu * mu;
  prm[i] = mu;
  prm[B_ * C_ + i] = rsqrtf(var + 1e-5f);
}
__global__ void resid_k(float* __restrict__ rin, const float* __restrict__ t2,
                        const float* __restrict__ prm) {
  size_t idx = (size_t)blockIdx.x * blockDim.x + threadIdx.x;
  if (idx >= (size_t)B_ * C_ * N_) return;
  int bc = (int)(idx / N_);
  float mu = prm[bc], rs = prm[B_ * C_ + bc];
  rin[idx] += (t2[idx] - mu) * rs;
}
__global__ void combine_k(float* __restrict__ out, const float* __restrict__ x,
                          const float* __restrict__ Mf, const float* __restrict__ rin,
                          const float* __restrict__ t2, const float* __restrict__ prm) {
  size_t idx = (size_t)blockIdx.x * blockDim.x + threadIdx.x;
  if (idx >= (size_t)B_ * C_ * N_) return;
  int nn = (int)(idx % N_);
  int bc = (int)(idx / N_);
  int b = bc / C_;
  float mu = prm[bc], rs = prm[B_ * C_ + bc];
  float wy = rin[idx] + (t2[idx] - mu) * rs;
  float m = Mf[(size_t)b * N_ + nn];
  out[idx] = m * x[idx] + (1.f - m) * wy;
}

// ---------------- launch ----------------
extern "C" void kernel_launch(void* const* d_in, const int* in_sizes, int n_in,
                              void* d_out, int out_size, void* d_ws, size_t ws_size,
                              hipStream_t stream) {
  (void)in_sizes; (void)n_in; (void)out_size; (void)ws_size;
  const float* x       = (const float*)d_in[0];
  const float* maskp   = (const float*)d_in[1];
  const float* g_w     = (const float*)d_in[2];
  const float* g_b     = (const float*)d_in[3];
  const float* theta_w = (const float*)d_in[4];
  const float* theta_b = (const float*)d_in[5];
  const float* phi_w   = (const float*)d_in[6];
  const float* phi_b   = (const float*)d_in[7];
  const float* W_w     = (const float*)d_in[8];
  const float* W_b     = (const float*)d_in[9];
  const float* res_w1  = (const float*)d_in[10];
  const float* res_b1  = (const float*)d_in[11];
  const float* res_w2  = (const float*)d_in[12];
  const float* res_b2  = (const float*)d_in[13];

  char* ws = (char*)d_ws;
  size_t off = 0;
  auto alloc = [&](size_t bytes) -> void* {
    void* p = ws + off; off += (bytes + 255) & ~(size_t)255; return p;
  };
  bf16* theta_bf = (bf16*)alloc((size_t)B_ * N_ * C_ * 2);
  bf16* phi_bf   = (bf16*)alloc((size_t)B_ * N_ * C_ * 2);
  bf16* g_bf     = (bf16*)alloc((size_t)B_ * N_ * C_ * 2);
  bf16* y_bf     = (bf16*)alloc((size_t)B_ * N_ * C_ * 2);
  float* Mf      = (float*)alloc((size_t)B_ * N_ * 4);
  float* r_in    = (float*)alloc((size_t)B_ * C_ * N_ * 4);
  float* t1      = (float*)alloc((size_t)B_ * C_ * N_ * 4);
  float* t2      = (float*)alloc((size_t)B_ * C_ * N_ * 4);
  float* stats   = (float*)alloc((size_t)2 * B_ * C_ * 4);
  float* pA      = (float*)alloc((size_t)2 * B_ * C_ * 4);
  float* pB      = (float*)alloc((size_t)2 * B_ * C_ * 4);

  mask_k<<<(B_ * N_ + 255) / 256, 256, 0, stream>>>(maskp, Mf);

  gemm_qkv<<<dim3(N_ / 64, C_ / 32, B_ * 3), 128, 0, stream>>>(
      x, theta_w, theta_b, phi_w, phi_b, g_w, g_b, theta_bf, phi_bf, g_bf);

  attn<<<dim3(N_ / 64, B_), 128, 0, stream>>>(theta_bf, phi_bf, g_bf, Mf, y_bf);

  gemm_w<<<dim3(N_ / 64, C_ / 32, B_), 128, 0, stream>>>(y_bf, W_w, W_b, r_in);

  for (int i = 0; i < 3; ++i) {
    const float* w1 = res_w1 + (size_t)i * C_ * C_ * 9;
    const float* b1 = res_b1 + i * C_;
    const float* w2 = res_w2 + (size_t)i * C_ * C_ * 9;
    const float* b2 = res_b2 + i * C_;

    zero_k<<<(2 * B_ * C_ + 255) / 256, 256, 0, stream>>>(stats, 2 * B_ * C_);
    conv3<<<dim3(128, C_ / 32, B_), 128, 0, stream>>>(r_in, pA, 0, w1, b1, t1, stats);
    finalize_k<<<(B_ * C_ + 255) / 256, 256, 0, stream>>>(stats, pA);

    zero_k<<<(2 * B_ * C_ + 255) / 256, 256, 0, stream>>>(stats, 2 * B_ * C_);
    conv3<<<dim3(128, C_ / 32, B_), 128, 0, stream>>>(t1, pA, 1, w2, b2, t2, stats);
    finalize_k<<<(B_ * C_ + 255) / 256, 256, 0, stream>>>(stats, pB);

    if (i < 2)
      resid_k<<<(int)(((size_t)B_ * C_ * N_) / 256), 256, 0, stream>>>(r_in, t2, pB);
  }

  combine_k<<<(int)(((size_t)B_ * C_ * N_) / 256), 256, 0, stream>>>(
      (float*)d_out, x, Mf, r_in, t2, pB);
}
